// MultiHeadAttention_64407329571337
// MI455X (gfx1250) — matrix operand order repaired and checked by reading, not compiled, not run on hardware
//
#include <hip/hip_runtime.h>

#define B_  2
#define T_  2048
#define D_  1024
#define H_  16
#define DH_ 64

typedef __bf16 bf16;
typedef __attribute__((ext_vector_type(8)))  __bf16 v8bf;
typedef __attribute__((ext_vector_type(16))) __bf16 v16bf;
typedef __attribute__((ext_vector_type(8)))  float  v8f;

#if __has_builtin(__builtin_amdgcn_tensor_load_to_lds) && \
    __has_builtin(__builtin_amdgcn_s_wait_tensorcnt)
#define USE_TDM 1
typedef __attribute__((ext_vector_type(4))) unsigned int u32x4;
typedef __attribute__((ext_vector_type(4))) int          i32x4;
typedef __attribute__((ext_vector_type(8))) int          i32x8;
#endif

// hardware RNE f32 -> bf16 convert (v_cvt_*bf16*), replaces manual bit-twiddling
static __device__ __forceinline__ bf16 f2bf(float f) { return (bf16)f; }

static __device__ __forceinline__ v16bf cat8(v8bf lo, v8bf hi) {
  return __builtin_shufflevector(lo, hi, 0,1,2,3,4,5,6,7,8,9,10,11,12,13,14,15);
}

static __device__ __forceinline__ v8f wmma_bf16(v16bf a, v16bf b, v8f c) {
  // D(16x16,f32) = A(16x32,bf16) * B(32x16,bf16) + C
  return __builtin_amdgcn_wmma_f32_16x16x32_bf16(false, a, false, b, (short)0, c,
                                                 false, false);
}

// ---------------------------------------------------------------------------
// Kernel 1: QKV projection.  C[4096,1024] = X @ W^T -> bf16 [B,H,T,Dh].
// Reference name swap: keys = x@Wq.T (used as K), queries = x@Wk.T (used as Q).
// BK = 64: 8 WMMA per wave per LDS round-trip.
// ---------------------------------------------------------------------------
__global__ __launch_bounds__(256) void qkv_proj_kernel(
    const float* __restrict__ x,
    const float* __restrict__ Wq, const float* __restrict__ Wk,
    const float* __restrict__ Wv,
    bf16* __restrict__ Qh, bf16* __restrict__ Kh, bf16* __restrict__ Vh)
{
  const int z = blockIdx.z;
  const float* W = (z == 0) ? Wk : (z == 1) ? Wq : Wv;
  bf16*       Out = (z == 0) ? Qh : (z == 1) ? Kh : Vh;

  const int m0 = blockIdx.x * 128;
  const int n0 = blockIdx.y * 64;
  const int tid  = threadIdx.x;
  const int wave = tid >> 5, lane = tid & 31;

  __shared__ __align__(16) bf16 As[128 * 72];   // stride 72 -> 144B rows, conflict free
  __shared__ __align__(16) bf16 Bs[64 * 72];

  v8f acc[4] = {};

  const int am = lane & 15, aklo = (lane >> 4) * 16;  // A-frag layout (16x32)
  const int bn = lane & 15, bklo = (lane >> 4) * 16;  // B-frag layout (32x16)

  for (int kk = 0; kk < D_; kk += 64) {
    // stage X tile 128x64 (f32 -> bf16)
    #pragma unroll
    for (int i = 0; i < 8; ++i) {
      int e = tid + i * 256;
      int r = e >> 4, c4 = e & 15;
      float4 v = *reinterpret_cast<const float4*>(x + (size_t)(m0 + r) * D_ + kk + c4 * 4);
      int o = r * 72 + c4 * 4;
      As[o+0] = f2bf(v.x); As[o+1] = f2bf(v.y); As[o+2] = f2bf(v.z); As[o+3] = f2bf(v.w);
    }
    // stage W tile 64x64 (f32 -> bf16)
    #pragma unroll
    for (int i = 0; i < 4; ++i) {
      int e = tid + i * 256;
      int r = e >> 4, c4 = e & 15;
      float4 v = *reinterpret_cast<const float4*>(W + (size_t)(n0 + r) * D_ + kk + c4 * 4);
      int o = r * 72 + c4 * 4;
      Bs[o+0] = f2bf(v.x); Bs[o+1] = f2bf(v.y); Bs[o+2] = f2bf(v.z); Bs[o+3] = f2bf(v.w);
    }
    if (kk + 64 < D_)  // warm L2/WGP$ for next K-slab (global_prefetch_b8)
      __builtin_prefetch(x + (size_t)(m0 + (tid >> 1)) * D_ + (kk + 64) + (tid & 1) * 32, 0, 0);
    __syncthreads();

    const v8bf* pa = reinterpret_cast<const v8bf*>(&As[(wave * 16 + am) * 72 + aklo]);
    v16bf a0 = cat8(pa[0], pa[1]);        // K = kk .. kk+31
    v16bf a1 = cat8(pa[4], pa[5]);        // K = kk+32 .. kk+63
    #pragma unroll
    for (int nt = 0; nt < 4; ++nt) {
      const v8bf* pb = reinterpret_cast<const v8bf*>(&Bs[(nt * 16 + bn) * 72 + bklo]);
      v16bf b0 = cat8(pb[0], pb[1]);
      v16bf b1 = cat8(pb[4], pb[5]);
      acc[nt] = wmma_bf16(a0, b0, acc[nt]);
      acc[nt] = wmma_bf16(a1, b1, acc[nt]);
    }
    __syncthreads();
  }

  // scatter C-frags into per-head [B,H,T,Dh] bf16
  const int mbase = m0 + wave * 16 + ((lane >> 4) << 3);
  #pragma unroll
  for (int nt = 0; nt < 4; ++nt) {
    const int n = n0 + nt * 16 + (lane & 15);
    const int h = n >> 6, d = n & 63;
    #pragma unroll
    for (int r = 0; r < 8; ++r) {
      const int m = mbase + r;
      const int b = m >> 11, t = m & (T_ - 1);
      Out[(((size_t)b * H_ + h) * T_ + t) * DH_ + d] = f2bf(acc[nt][r]);
    }
  }
}

// ---------------------------------------------------------------------------
// Kernel 2: causal flash attention.  One workgroup = (b,h) x 128 query rows.
// ---------------------------------------------------------------------------
__global__ __launch_bounds__(256) void attn_kernel(
    const bf16* __restrict__ Qh, const bf16* __restrict__ Kh,
    const bf16* __restrict__ Vh, bf16* __restrict__ ctx)
{
  const int qt  = blockIdx.x;      // 0..15 query tile
  const int bh  = blockIdx.y;      // 0..31 (b*16+h)
  const int tid = threadIdx.x;
  const int wave = tid >> 5, lane = tid & 31;

  const size_t base = (size_t)bh * T_ * DH_;
  const int q0  = qt * 128;
  const int qr0 = q0 + wave * 16;

  __shared__ __align__(16) bf16 Vt[DH_ * 136];     // V transposed: [d][key]
  __shared__ __align__(16) bf16 Ps[8][16 * 136];   // per-wave P re-layout buffer

  const int am   = lane & 15;
  const int half = lane >> 4;
  const int aklo = half * 16;

  // resident Q A-fragments (Dh = 64 -> two 16x32 frags)
  const v8bf* qp = reinterpret_cast<const v8bf*>(Qh + base + (size_t)(qr0 + am) * DH_);
  const int qi = aklo >> 3;                        // 0 or 2
  v16bf aq0 = cat8(qp[qi],     qp[qi + 1]);
  v16bf aq1 = cat8(qp[qi + 4], qp[qi + 5]);

  v8f acc[4] = {};
  float mstat[8], lstat[8];
  #pragma unroll
  for (int r = 0; r < 8; ++r) { mstat[r] = -3.0e38f; lstat[r] = 0.0f; }

  const float scale = 0.125f;                      // 1/sqrt(64)

  for (int j = 0; j <= qt; ++j) {                  // causal: key tiles <= query tile
    const int kbase = j * 128;

    // stage V tile transposed into LDS
    #pragma unroll
    for (int i = 0; i < 4; ++i) {
      int e = tid + i * 256;                       // 1024 chunks of 8 bf16
      int r = e >> 3, c8 = e & 7;
      v8bf v = *reinterpret_cast<const v8bf*>(Vh + base + (size_t)(kbase + r) * DH_ + c8 * 8);
      #pragma unroll
      for (int u = 0; u < 8; ++u) Vt[(c8 * 8 + u) * 136 + r] = v[u];
    }
    __syncthreads();

    // S(16x128) = Q K^T : K rows are exactly B-frag columns (contiguous loads)
    v8f s[8];
    #pragma unroll
    for (int ct = 0; ct < 8; ++ct) {
      v8f sf = {};
      const int krow = kbase + ct * 16 + am;
      const v8bf* kp = reinterpret_cast<const v8bf*>(Kh + base + (size_t)krow * DH_);
      const int di = half * 2;
      v16bf b0 = cat8(kp[di],     kp[di + 1]);     // d in [half*16, +16)
      v16bf b1 = cat8(kp[di + 4], kp[di + 5]);     // d in [32+half*16, +16)
      sf = wmma_bf16(aq0, b0, sf);
      sf = wmma_bf16(aq1, b1, sf);
      s[ct] = sf;
    }

    // scale + causal mask (diagonal tile only)
    const bool diag = (j == qt);
    #pragma unroll
    for (int ct = 0; ct < 8; ++ct) {
      const int key = kbase + ct * 16 + am;
      #pragma unroll
      for (int r = 0; r < 8; ++r) {
        float v = s[ct][r] * scale;
        if (diag) {
          const int qrow = qr0 + r + (half << 3);
          if (key > qrow) v = -3.0e38f;
        }
        s[ct][r] = v;
      }
    }

    // online softmax: row stats live across 16-lane halves (C-frag layout)
    #pragma unroll
    for (int r = 0; r < 8; ++r) {
      float vm = s[0][r];
      #pragma unroll
      for (int ct = 1; ct < 8; ++ct) vm = fmaxf(vm, s[ct][r]);
      vm = fmaxf(vm, __shfl_xor(vm, 1, 32));
      vm = fmaxf(vm, __shfl_xor(vm, 2, 32));
      vm = fmaxf(vm, __shfl_xor(vm, 4, 32));
      vm = fmaxf(vm, __shfl_xor(vm, 8, 32));
      const float nm   = fmaxf(mstat[r], vm);
      const float corr = __expf(mstat[r] - nm);
      mstat[r] = nm;
      lstat[r] *= corr;
      #pragma unroll
      for (int f = 0; f < 4; ++f) acc[f][r] *= corr;
    }
    #pragma unroll
    for (int ct = 0; ct < 8; ++ct) {
      #pragma unroll
      for (int r = 0; r < 8; ++r) s[ct][r] = __expf(s[ct][r] - mstat[r]);
    }
    #pragma unroll
    for (int r = 0; r < 8; ++r) {
      float sum = 0.0f;
      #pragma unroll
      for (int ct = 0; ct < 8; ++ct) sum += s[ct][r];
      sum += __shfl_xor(sum, 1, 32);
      sum += __shfl_xor(sum, 2, 32);
      sum += __shfl_xor(sum, 4, 32);
      sum += __shfl_xor(sum, 8, 32);
      lstat[r] += sum;
    }

    // C-frag -> A-frag re-layout of P through this wave's LDS buffer
    bf16* P = &Ps[wave][0];
    #pragma unroll
    for (int ct = 0; ct < 8; ++ct) {
      #pragma unroll
      for (int r = 0; r < 8; ++r)
        P[(r + (half << 3)) * 136 + ct * 16 + am] = f2bf(s[ct][r]);
    }
    // same-wave DS ops are in-order: no barrier needed before re-reading P

    // acc(16x64) += P(16x128) @ V(128x64)
    #pragma unroll
    for (int ks = 0; ks < 4; ++ks) {
      const int kk0 = ks * 32;
      const v8bf* pp = reinterpret_cast<const v8bf*>(&P[am * 136 + kk0 + aklo]);
      v16bf ap = cat8(pp[0], pp[1]);
      #pragma unroll
      for (int nt = 0; nt < 4; ++nt) {
        const v8bf* vp =
            reinterpret_cast<const v8bf*>(&Vt[(nt * 16 + am) * 136 + kk0 + half * 16]);
        v16bf bv = cat8(vp[0], vp[1]);
        acc[nt] = wmma_bf16(ap, bv, acc[nt]);
      }
    }
    __syncthreads();
  }

  // normalize and store ctx as bf16 [B,T,D] (head-concat layout)
  const int b = bh >> 4, h = bh & 15;
  #pragma unroll
  for (int r = 0; r < 8; ++r) {
    const float inv = 1.0f / lstat[r];
    const int t = qr0 + r + (half << 3);
    #pragma unroll
    for (int nt = 0; nt < 4; ++nt) {
      const int d = nt * 16 + am;
      ctx[((size_t)b * T_ + t) * D_ + h * DH_ + d] = f2bf(acc[nt][r] * inv);
    }
  }
}

// ---------------------------------------------------------------------------
// Kernel 3: output projection.  out[4096,1024] = ctx(bf16) @ Wo^T + bo  (f32)
// A-tile staged by the Tensor Data Mover (pure bf16 2D tile copy) when the
// TDM builtin is available; TDM pad feature reproduces the stride-72 layout.
// ---------------------------------------------------------------------------
__global__ __launch_bounds__(256) void out_proj_kernel(
    const bf16* __restrict__ ctxm, const float* __restrict__ Wo,
    const float* __restrict__ bo, float* __restrict__ out)
{
  const int m0 = blockIdx.x * 128;
  const int n0 = blockIdx.y * 64;
  const int tid  = threadIdx.x;
  const int wave = tid >> 5, lane = tid & 31;

  __shared__ __align__(16) bf16 As[128 * 72];
  __shared__ __align__(16) bf16 Bs[64 * 72];

  v8f acc[4] = {};
  const int am = lane & 15, aklo = (lane >> 4) * 16;
  const int bn = lane & 15, bklo = (lane >> 4) * 16;

  for (int kk = 0; kk < D_; kk += 64) {
#ifdef USE_TDM
    if (tid < 32) {  // one wave issues the tile DMA (EXEC ignored by TDM)
      const unsigned long long ga =
          (unsigned long long)(uintptr_t)ctxm + ((size_t)m0 * D_ + kk) * 2ull;
      const unsigned lds = (unsigned)(uintptr_t)&As[0];
      u32x4 g0;
      g0[0] = 1u;                                        // count=1, user descriptor
      g0[1] = lds;                                       // lds_addr (bytes)
      g0[2] = (unsigned)(ga & 0xFFFFFFFFull);            // global_addr[31:0]
      g0[3] = (unsigned)((ga >> 32) & 0x1FFFFFFull)      // global_addr[56:32]
            | (2u << 30);                                // type = 2 ("image")
      i32x8 g1;
      g1[0] = (1 << 16)                                  // data_size = 1 (2 bytes)
            | (1 << 20)                                  // pad_enable
            | (4 << 22)                                  // pad_interval: 32 dwords (row)
            | (3 << 25);                                 // pad_amount: 4 dwords -> stride 72
      g1[1] = (int)((D_ & 0xFFFF) << 16);                // tensor_dim0 = 1024 (lo16)
      g1[2] = (int)((D_ >> 16) | ((4096 & 0xFFFF) << 16));// dim0 hi16 | tensor_dim1=4096 lo16
      g1[3] = (int)((4096 >> 16) | (64 << 16));          // dim1 hi16 | tile_dim0 = 64
      g1[4] = 128;                                       // tile_dim1 = 128, tile_dim2 = 0
      g1[5] = D_;                                        // tensor_dim0_stride = 1024 (lo32)
      g1[6] = 0;                                         // stride0 hi16 | stride1 lo16
      g1[7] = 0;
      i32x4 gz = {0, 0, 0, 0};
#if defined(__clang_major__) && __clang_major__ >= 23
      i32x8 gz8 = {0, 0, 0, 0, 0, 0, 0, 0};
      __builtin_amdgcn_tensor_load_to_lds(g0, g1, gz, gz, gz8, 0);
#else
      __builtin_amdgcn_tensor_load_to_lds(g0, g1, gz, gz, 0);
#endif
      __builtin_amdgcn_s_wait_tensorcnt(0);
    }
#else
    #pragma unroll
    for (int i = 0; i < 4; ++i) {
      int e = tid + i * 256;                 // 1024 chunks of 8 bf16
      int r = e >> 3, c8 = e & 7;
      v8bf v = *reinterpret_cast<const v8bf*>(ctxm + (size_t)(m0 + r) * D_ + kk + c8 * 8);
      *reinterpret_cast<v8bf*>(&As[r * 72 + c8 * 8]) = v;
    }
#endif
    // stage Wo tile 64x64 (f32 -> bf16)
    #pragma unroll
    for (int i = 0; i < 4; ++i) {
      int e = tid + i * 256;
      int r = e >> 4, c4 = e & 15;
      float4 v = *reinterpret_cast<const float4*>(Wo + (size_t)(n0 + r) * D_ + kk + c4 * 4);
      int o = r * 72 + c4 * 4;
      Bs[o+0] = f2bf(v.x); Bs[o+1] = f2bf(v.y); Bs[o+2] = f2bf(v.z); Bs[o+3] = f2bf(v.w);
    }
    __syncthreads();

    const v8bf* pa = reinterpret_cast<const v8bf*>(&As[(wave * 16 + am) * 72 + aklo]);
    v16bf a0 = cat8(pa[0], pa[1]);
    v16bf a1 = cat8(pa[4], pa[5]);
    #pragma unroll
    for (int nt = 0; nt < 4; ++nt) {
      const v8bf* pb = reinterpret_cast<const v8bf*>(&Bs[(nt * 16 + bn) * 72 + bklo]);
      v16bf b0 = cat8(pb[0], pb[1]);
      v16bf b1 = cat8(pb[4], pb[5]);
      acc[nt] = wmma_bf16(a0, b0, acc[nt]);
      acc[nt] = wmma_bf16(a1, b1, acc[nt]);
    }
    __syncthreads();
  }

  const int mbase = m0 + wave * 16 + ((lane >> 4) << 3);
  #pragma unroll
  for (int nt = 0; nt < 4; ++nt) {
    const int n = n0 + nt * 16 + (lane & 15);
    const float bias = bo[n];
    #pragma unroll
    for (int r = 0; r < 8; ++r)
      out[(size_t)(mbase + r) * D_ + n] = acc[nt][r] + bias;
  }
}

// ---------------------------------------------------------------------------
extern "C" void kernel_launch(void* const* d_in, const int* in_sizes, int n_in,
                              void* d_out, int out_size, void* d_ws, size_t ws_size,
                              hipStream_t stream) {
  (void)in_sizes; (void)n_in; (void)out_size; (void)ws_size;
  const float* x  = (const float*)d_in[0];
  const float* Wq = (const float*)d_in[1];
  const float* Wk = (const float*)d_in[2];
  const float* Wv = (const float*)d_in[3];
  const float* Wo = (const float*)d_in[4];
  const float* bo = (const float*)d_in[5];
  float* out = (float*)d_out;

  const size_t per = (size_t)B_ * H_ * T_ * DH_;   // 4,194,304 elems
  bf16* Qh  = (bf16*)d_ws;
  bf16* Kh  = Qh + per;
  bf16* Vh  = Kh + per;
  bf16* ctx = Vh + per;                            // total ~33.5 MB bf16 scratch

  qkv_proj_kernel<<<dim3(32, 16, 3), 256, 0, stream>>>(x, Wq, Wk, Wv, Qh, Kh, Vh);
  attn_kernel<<<dim3(16, 32), 256, 0, stream>>>(Qh, Kh, Vh, ctx);
  out_proj_kernel<<<dim3(32, 16), 256, 0, stream>>>(ctx, Wo, bo, out);
}
